// AST_GODE_88983132438674
// MI455X (gfx1250) — compile-verified, matrix operand order checked by reading, not run on hardware
//
#include <hip/hip_runtime.h>

#define DEV __device__ __forceinline__

typedef __attribute__((ext_vector_type(16))) __bf16        v16bf;
typedef __attribute__((ext_vector_type(8)))  float         v8f;
typedef __attribute__((ext_vector_type(4)))  unsigned int  u32x4;

union Frag {
  v16bf          v;
  u32x4          q[2];
  unsigned short u[16];
};

constexpr int BB    = 16;    // batch
constexpr int NV    = 2048;  // nodes
constexpr int FF    = 2;     // features
constexpr int TIN   = 12;    // input time steps
constexpr int HH    = 64;    // hidden
constexpr int TSV   = 65;    // H+1
constexpr int TSP   = 96;    // TS padded to 3x32 for WMMA K
constexpr int NSTEP = 12;    // NEVAL-1

DEV unsigned short f2bf(float f) {
  unsigned int u = __float_as_uint(f);
  unsigned int r = u + 0x7FFFu + ((u >> 16) & 1u);
  return (unsigned short)(r >> 16);
}
DEV u32x4 ld128(const unsigned short* p) { return *(const u32x4*)p; }
DEV v8f vzero() { v8f z = {0.f,0.f,0.f,0.f,0.f,0.f,0.f,0.f}; return z; }

// ---------- one-time prep: bf16 copies of adj (row & col major) and Vs ----------
__global__ __launch_bounds__(256) void prep_bf(const float* __restrict__ adj,
                                               const float* __restrict__ Vs,
                                               unsigned short* __restrict__ adj_bf,
                                               unsigned short* __restrict__ adjT_bf,
                                               unsigned short* __restrict__ Vs_bf) {
  int idx = blockIdx.x * 256 + threadIdx.x;
  if (idx >= NV * NV) return;
  int r = idx >> 11, c = idx & (NV - 1);
  unsigned short a = f2bf(adj[idx]);
  adj_bf[idx] = a;
  adjT_bf[(size_t)c * NV + r] = a;
  Vs_bf[idx] = f2bf(Vs[idx]);
}

__global__ void prep_scal(const float* __restrict__ W1, const float* __restrict__ W3,
                          float* __restrict__ scal) {
  if (threadIdx.x == 0 && blockIdx.x == 0) {
    float s = 0.f;
    for (int t = 0; t < TSV; ++t) s += W1[t];
    scal[0] = s;
    scal[1] = W3[0] + W3[1];
  }
}

// ---------- T2 = 2*adj@adj - I via bf16 WMMA (one 16x16 tile per wave) ----------
__global__ __launch_bounds__(32) void t2_gemm(const unsigned short* __restrict__ adj_bf,
                                              const unsigned short* __restrict__ adjT_bf,
                                              float* __restrict__ T2) {
  const int i0 = blockIdx.x * 16, j0 = blockIdx.y * 16;
  const int lane = threadIdx.x & 31, hi = lane >> 4, lcol = lane & 15;
  v8f acc = vzero();
  const unsigned short* arow = adj_bf  + (size_t)(i0 + lcol) * NV;
  const unsigned short* bcol = adjT_bf + (size_t)(j0 + lcol) * NV;
  for (int k0 = 0; k0 < NV; k0 += 32) {
    Frag a, b;
    a.q[0] = ld128(arow + k0 + hi * 8);  a.q[1] = ld128(arow + k0 + 16 + hi * 8);
    b.q[0] = ld128(bcol + k0 + hi * 8);  b.q[1] = ld128(bcol + k0 + 16 + hi * 8);
    acc = __builtin_amdgcn_wmma_f32_16x16x32_bf16(false, a.v, false, b.v, (short)0, acc, false, false);
  }
#pragma unroll
  for (int j = 0; j < 8; ++j) {
    int row = i0 + j + 8 * hi, col = j0 + lcol;
    T2[(size_t)row * NV + col] = 2.0f * acc[j] - (row == col ? 1.0f : 0.0f);
  }
}

// ---------- gc = relu(einsum(knm,bnft,kfg->bmgt)) ; VALU (only ~19 GFLOP once) ----------
__global__ __launch_bounds__(256) void gc_kernel(const float* __restrict__ x,
                                                 const float* __restrict__ adj,
                                                 const float* __restrict__ T2,
                                                 const float* __restrict__ theta_init,
                                                 float* __restrict__ gc) {
  const int m = blockIdx.x * 256 + threadIdx.x;
  const int b = blockIdx.y;
  __shared__ float xs[16 * FF * TIN];  // 384 floats
  float u1[FF][TIN], u2[FF][TIN];
#pragma unroll
  for (int f = 0; f < FF; ++f)
#pragma unroll
    for (int t = 0; t < TIN; ++t) { u1[f][t] = 0.f; u2[f][t] = 0.f; }

  for (int nb = 0; nb < NV; nb += 16) {
    __syncthreads();
    for (int i = threadIdx.x; i < 16 * FF * TIN; i += 256)
      xs[i] = x[((size_t)b * NV + nb) * (FF * TIN) + i];
    __syncthreads();
    for (int nn = 0; nn < 16; ++nn) {
      int n = nb + nn;
      float av = adj[(size_t)n * NV + m];
      float a2 = T2[(size_t)n * NV + m];
#pragma unroll
      for (int f = 0; f < FF; ++f)
#pragma unroll
        for (int t = 0; t < TIN; ++t) {
          float xv = xs[nn * (FF * TIN) + f * TIN + t];
          u1[f][t] += av * xv;
          u2[f][t] += a2 * xv;
        }
    }
  }
  float th[3][2][2];
#pragma unroll
  for (int k = 0; k < 3; ++k)
#pragma unroll
    for (int f = 0; f < 2; ++f)
#pragma unroll
      for (int g = 0; g < 2; ++g) th[k][f][g] = theta_init[(k * 2 + f) * 2 + g];
  float xm[2][TIN];
#pragma unroll
  for (int f = 0; f < 2; ++f)
#pragma unroll
    for (int t = 0; t < TIN; ++t) xm[f][t] = x[(((size_t)b * NV + m) * 2 + f) * TIN + t];
#pragma unroll
  for (int g = 0; g < 2; ++g)
#pragma unroll
    for (int t = 0; t < TIN; ++t) {
      float v = th[0][0][g] * xm[0][t] + th[0][1][g] * xm[1][t]
              + th[1][0][g] * u1[0][t] + th[1][1][g] * u1[1][t]
              + th[2][0][g] * u2[0][t] + th[2][1][g] * u2[1][t];
      gc[(((size_t)b * NV + m) * 2 + g) * TIN + t] = fmaxf(v, 0.f);
    }
}

// ---------- h0 = relu(gc@w_t1+b1)@w_t2+b2 ; plus sum_h h0 and dot(h0, W1[1:]) ----------
__global__ __launch_bounds__(64) void h0_kernel(const float* __restrict__ gc,
                                                const float* __restrict__ w_t1,
                                                const float* __restrict__ b_t1,
                                                const float* __restrict__ w_t2,
                                                const float* __restrict__ b_t2,
                                                const float* __restrict__ W1,
                                                float* __restrict__ h0,
                                                float* __restrict__ h0sum,
                                                float* __restrict__ h0w1) {
  const int bnf = blockIdx.x;
  const int tid = threadIdx.x;
  __shared__ float g12[TIN], hl[HH], red[HH];
  if (tid < TIN) g12[tid] = gc[(size_t)bnf * TIN + tid];
  __syncthreads();
  float hv = b_t1[tid];
#pragma unroll
  for (int t = 0; t < TIN; ++t) hv += g12[t] * w_t1[t * HH + tid];
  hv = fmaxf(hv, 0.f);
  hl[tid] = hv;
  __syncthreads();
  float ov = b_t2[tid];
  for (int d = 0; d < HH; ++d) ov += hl[d] * w_t2[d * HH + tid];
  h0[(size_t)bnf * HH + tid] = ov;

  red[tid] = ov; __syncthreads();
  for (int s = 32; s > 0; s >>= 1) { if (tid < s) red[tid] += red[tid + s]; __syncthreads(); }
  if (tid == 0) h0sum[bnf] = red[0];
  __syncthreads();
  red[tid] = ov * W1[1 + tid]; __syncthreads();
  for (int s = 32; s > 0; s >>= 1) { if (tid < s) red[tid] += red[tid + s]; __syncthreads(); }
  if (tid == 0) h0w1[bnf] = red[0];
}

__global__ __launch_bounds__(256) void y0_kernel(const float* __restrict__ x,
                                                 float* __restrict__ y0b) {
  int i = blockIdx.x * 256 + threadIdx.x;
  if (i < BB * NV * FF) y0b[i] = x[(size_t)i * TIN + (TIN - 1)];
}

// ---------- per step: lhs/rhs (bf16 padded K=96) and xsum from current y ----------
__global__ __launch_bounds__(96) void stepA(int s, const float* __restrict__ t_span,
                                            const float* __restrict__ w_te,
                                            const float* __restrict__ b_te,
                                            const float* __restrict__ W1,
                                            const float* __restrict__ W2,
                                            const float* __restrict__ W3,
                                            const float* __restrict__ scal,
                                            const float* __restrict__ y0b,
                                            const float* __restrict__ dout,
                                            const float* __restrict__ h0,
                                            const float* __restrict__ h0sum,
                                            const float* __restrict__ h0w1,
                                            unsigned short* __restrict__ lhs_bf,
                                            unsigned short* __restrict__ rhs_bf,
                                            float* __restrict__ xsum) {
  const int bn = blockIdx.x;
  const int n = bn & (NV - 1);
  const int tid = threadIdx.x;
  const float t = t_span[s];
  const float te = t * w_te[n] + b_te[n];
  const float* yp = (s == 0) ? y0b : (dout + (size_t)(s - 1) * BB * NV * FF);
  const float yv0 = yp[(size_t)bn * 2 + 0];
  const float yv1 = yp[(size_t)bn * 2 + 1];
  if (tid == 0) xsum[(size_t)bn * 2 + 0] = yv0 + h0sum[(size_t)bn * 2 + 0] + (float)TSV * te;
  if (tid == 1) xsum[(size_t)bn * 2 + 1] = yv1 + h0sum[(size_t)bn * 2 + 1] + (float)TSV * te;
  float lv = 0.f, rv = 0.f;
  if (tid < TSV) {
    const float W1sum = scal[0], W3sum = scal[1];
    const float a10 = yv0 * W1[0] + h0w1[(size_t)bn * 2 + 0] + te * W1sum;
    const float a11 = yv1 * W1[0] + h0w1[(size_t)bn * 2 + 1] + te * W1sum;
    lv = a10 * W2[tid] + a11 * W2[TSV + tid];
    if (tid == 0)
      rv = W3[0] * (yv0 + te) + W3[1] * (yv1 + te);
    else
      rv = W3[0] * h0[((size_t)bn * 2 + 0) * HH + tid - 1]
         + W3[1] * h0[((size_t)bn * 2 + 1) * HH + tid - 1] + te * W3sum;
  }
  lhs_bf[(size_t)bn * TSP + tid] = f2bf(lv);
  rhs_bf[(size_t)bn * TSP + tid] = f2bf(rv);
}

// ---------- per step: fused prod->sigmoid->Vs GEMM->softmax->final contraction ----------
__global__ __launch_bounds__(256) void stepB(int s, const float* __restrict__ t_span,
                                             const unsigned short* __restrict__ lhs_bf,
                                             const unsigned short* __restrict__ rhs_bf,
                                             const unsigned short* __restrict__ Vs_bf,
                                             const float* __restrict__ bs,
                                             const float* __restrict__ adj,
                                             const float* __restrict__ T2,
                                             const float* __restrict__ xsum,
                                             const float* __restrict__ theta_att,
                                             const float* __restrict__ y0b,
                                             float* __restrict__ dout) {
  const int b    = blockIdx.y;
  const int m0   = blockIdx.x * 16;
  const int tid  = threadIdx.x;
  const int wave = tid >> 5;
  const int lane = tid & 31;
  const int hi   = lane >> 4;
  const int lcol = lane & 15;

  __shared__ float red[256];
  __shared__ float colmax[16];
  __shared__ float colZ[16];
  __shared__ float Pm[16][6];
  __shared__ float xsl[8][512];

  // rhs B-fragments (K=96 -> 3 fragments), constant over the k loop
  Frag rB[3];
  {
    const unsigned short* rbase = rhs_bf + (size_t)(b * NV + m0 + lcol) * TSP;
#pragma unroll
    for (int ks = 0; ks < 3; ++ks) {
      rB[ks].q[0] = ld128(rbase + ks * 32 + hi * 8);
      rB[ks].q[1] = ld128(rbase + ks * 32 + 16 + hi * 8);
    }
  }

  v8f acc[16];
#pragma unroll
  for (int t = 0; t < 16; ++t) acc[t] = vzero();
  if (tid < 96) ((float*)Pm)[tid] = 0.f;

  for (int k0 = 0; k0 < NV; k0 += 32) {
    // prod tile (32 k-rows x 16 m-cols), redundantly per wave, K=96
    v8f p0 = vzero(), p1 = vzero();
    const unsigned short* l0 = lhs_bf + (size_t)(b * NV + k0 + lcol) * TSP;
    const unsigned short* l1 = l0 + 16 * TSP;
#pragma unroll
    for (int ks = 0; ks < 3; ++ks) {
      Frag a0, a1;
      a0.q[0] = ld128(l0 + ks * 32 + hi * 8);  a0.q[1] = ld128(l0 + ks * 32 + 16 + hi * 8);
      a1.q[0] = ld128(l1 + ks * 32 + hi * 8);  a1.q[1] = ld128(l1 + ks * 32 + 16 + hi * 8);
      p0 = __builtin_amdgcn_wmma_f32_16x16x32_bf16(false, a0.v, false, rB[ks].v, (short)0, p0, false, false);
      p1 = __builtin_amdgcn_wmma_f32_16x16x32_bf16(false, a1.v, false, rB[ks].v, (short)0, p1, false, false);
    }
    // sigmoid(prod + bs); C-layout maps lane-locally onto the B-fragment layout
    float sg[16];
#pragma unroll
    for (int j = 0; j < 8; ++j) {
      const int kk0 = k0 + j + 8 * hi;
      const float v0 = p0[j] + bs[(size_t)kk0 * NV + m0 + lcol];
      const float v1 = p1[j] + bs[(size_t)(kk0 + 16) * NV + m0 + lcol];
      sg[j]     = 1.0f / (1.0f + __expf(-v0));
      sg[8 + j] = 1.0f / (1.0f + __expf(-v1));
    }
    Frag sf;
#pragma unroll
    for (int v = 0; v < 8; ++v) {
      const int t = v >> 2, j = (v & 3) * 2;
      sf.u[2 * v]     = f2bf(sg[t * 8 + j]);
      sf.u[2 * v + 1] = f2bf(sg[t * 8 + j + 1]);
    }
    if (k0 + 32 < NV)
      __builtin_prefetch(lhs_bf + (size_t)(b * NV + k0 + 32 + lcol) * TSP, 0, 3);
    // main GEMM: S[n, m-tile] += Vs[n, k-chunk] @ sig ; wave owns 256 n rows
#pragma unroll
    for (int t = 0; t < 16; ++t) {
      const int row = wave * 256 + t * 16 + lcol;
      const unsigned short* vb = Vs_bf + (size_t)row * NV + k0;
      Frag a;
      a.q[0] = ld128(vb + hi * 8);
      a.q[1] = ld128(vb + 16 + hi * 8);
      acc[t] = __builtin_amdgcn_wmma_f32_16x16x32_bf16(false, a.v, false, sf.v, (short)0, acc[t], false, false);
    }
  }

  // ---- epilogue: softmax over n (axis=1) + final knm,bnm,bnf,kfg->bmg ----
  for (int i = lane; i < 512; i += 32)
    xsl[wave][i] = xsum[((size_t)(b * NV + wave * 256)) * 2 + i];

  float mymax = -3.4e38f;
#pragma unroll
  for (int t = 0; t < 16; ++t)
#pragma unroll
    for (int j = 0; j < 8; ++j) mymax = fmaxf(mymax, acc[t][j]);
  red[tid] = mymax;
  __syncthreads();
  if (tid < 16) {
    float mm = -3.4e38f;
    for (int i = 0; i < 16; ++i) mm = fmaxf(mm, red[i * 16 + tid]);
    colmax[tid] = mm;
  }
  __syncthreads();
  const float cm = colmax[lcol];
  float zpart = 0.f;
#pragma unroll
  for (int t = 0; t < 16; ++t)
#pragma unroll
    for (int j = 0; j < 8; ++j) {
      float e = __expf(acc[t][j] - cm);
      acc[t][j] = e;
      zpart += e;
    }
  red[tid] = zpart;
  __syncthreads();
  if (tid < 16) {
    float zz = 0.f;
    for (int i = 0; i < 16; ++i) zz += red[i * 16 + tid];
    colZ[tid] = zz;
  }
  __syncthreads();

  float p00 = 0, p01 = 0, p10 = 0, p11 = 0, p20 = 0, p21 = 0;
  const int mg = m0 + lcol;
#pragma unroll
  for (int t = 0; t < 16; ++t)
#pragma unroll
    for (int j = 0; j < 8; ++j) {
      const int n = wave * 256 + t * 16 + j + 8 * hi;
      const float e  = acc[t][j];
      const float c1 = adj[(size_t)n * NV + mg];
      const float c2 = T2[(size_t)n * NV + mg];
      const float xs0 = xsl[wave][(n - wave * 256) * 2 + 0];
      const float xs1 = xsl[wave][(n - wave * 256) * 2 + 1];
      p10 += e * c1 * xs0;  p11 += e * c1 * xs1;
      p20 += e * c2 * xs0;  p21 += e * c2 * xs1;
      if (n == mg) { p00 += e * xs0; p01 += e * xs1; }
    }
  atomicAdd(&Pm[lcol][0], p00);  atomicAdd(&Pm[lcol][1], p01);
  atomicAdd(&Pm[lcol][2], p10);  atomicAdd(&Pm[lcol][3], p11);
  atomicAdd(&Pm[lcol][4], p20);  atomicAdd(&Pm[lcol][5], p21);
  __syncthreads();

  if (tid < 32) {
    const int m = tid & 15, g = tid >> 4;
    float v = 0.f;
#pragma unroll
    for (int k = 0; k < 3; ++k)
#pragma unroll
      for (int f = 0; f < 2; ++f)
        v += Pm[m][k * 2 + f] * theta_att[(k * 2 + f) * 2 + g];
    v = fmaxf(v / colZ[m], 0.f);  // softmax denominator factored out
    const int mgx = m0 + m;
    const size_t oidx = ((size_t)b * NV + mgx) * 2 + g;
    const float* yp = (s == 0) ? y0b : (dout + (size_t)(s - 1) * BB * NV * FF);
    const float dt = t_span[s + 1] - t_span[s];
    dout[(size_t)s * BB * NV * FF + oidx] = yp[oidx] + dt * v;
  }
}

extern "C" void kernel_launch(void* const* d_in, const int* in_sizes, int n_in,
                              void* d_out, int out_size, void* d_ws, size_t ws_size,
                              hipStream_t stream) {
  const float* x    = (const float*)d_in[0];
  // d_in[1] = T (unused by the reference computation)
  const float* tspan= (const float*)d_in[2];
  const float* adj  = (const float*)d_in[3];
  const float* th_i = (const float*)d_in[4];
  const float* w_t1 = (const float*)d_in[5];
  const float* b_t1 = (const float*)d_in[6];
  const float* w_t2 = (const float*)d_in[7];
  const float* b_t2 = (const float*)d_in[8];
  const float* w_te = (const float*)d_in[9];
  const float* b_te = (const float*)d_in[10];
  const float* W1   = (const float*)d_in[11];
  const float* W2   = (const float*)d_in[12];
  const float* W3   = (const float*)d_in[13];
  const float* bsp  = (const float*)d_in[14];
  const float* Vs   = (const float*)d_in[15];
  const float* th_a = (const float*)d_in[16];
  float* dout = (float*)d_out;

  char* w = (char*)d_ws;
  auto alloc = [&](size_t bytes) -> char* {
    char* p = w;
    w += (bytes + 255) & ~(size_t)255;
    return p;
  };
  float*          T2      = (float*)alloc(sizeof(float) * NV * NV);
  unsigned short* adj_bf  = (unsigned short*)alloc(2ull * NV * NV);
  unsigned short* adjT_bf = (unsigned short*)alloc(2ull * NV * NV);
  unsigned short* Vs_bf   = (unsigned short*)alloc(2ull * NV * NV);
  float*          gc      = (float*)alloc(sizeof(float) * BB * NV * FF * TIN);
  float*          h0      = (float*)alloc(sizeof(float) * BB * NV * FF * HH);
  float*          h0s     = (float*)alloc(sizeof(float) * BB * NV * FF);
  float*          h0w     = (float*)alloc(sizeof(float) * BB * NV * FF);
  float*          xsumb   = (float*)alloc(sizeof(float) * BB * NV * FF);
  unsigned short* lhs_bf  = (unsigned short*)alloc(2ull * BB * NV * TSP);
  unsigned short* rhs_bf  = (unsigned short*)alloc(2ull * BB * NV * TSP);
  float*          y0b     = (float*)alloc(sizeof(float) * BB * NV * FF);
  float*          scal    = (float*)alloc(256);

  prep_bf  <<<(NV * NV + 255) / 256, 256, 0, stream>>>(adj, Vs, adj_bf, adjT_bf, Vs_bf);
  prep_scal<<<1, 32, 0, stream>>>(W1, W3, scal);
  t2_gemm  <<<dim3(NV / 16, NV / 16), 32, 0, stream>>>(adj_bf, adjT_bf, T2);
  gc_kernel<<<dim3(NV / 256, BB), 256, 0, stream>>>(x, adj, T2, th_i, gc);
  h0_kernel<<<BB * NV * FF, HH, 0, stream>>>(gc, w_t1, b_t1, w_t2, b_t2, W1, h0, h0s, h0w);
  y0_kernel<<<(BB * NV * FF + 255) / 256, 256, 0, stream>>>(x, y0b);

  for (int s = 0; s < NSTEP; ++s) {
    stepA<<<BB * NV, 96, 0, stream>>>(s, tspan, w_te, b_te, W1, W2, W3, scal,
                                      y0b, dout, h0, h0s, h0w, lhs_bf, rhs_bf, xsumb);
    stepB<<<dim3(NV / 16, BB), 256, 0, stream>>>(s, tspan, lhs_bf, rhs_bf, Vs_bf,
                                                 bsp, adj, T2, xsumb, th_a, y0b, dout);
  }
  (void)in_sizes; (void)n_in; (void)out_size; (void)ws_size;
}
